// RNNAdder_39891656245930
// MI455X (gfx1250) — matrix operand order repaired and checked by reading, not compiled, not run on hardware
//
#include <hip/hip_runtime.h>
#include <stdint.h>
#include <math.h>

typedef __attribute__((ext_vector_type(16))) _Float16 v16h;
typedef __attribute__((ext_vector_type(8)))  float    v8f;

#define VOCAB   10
#define EMB     32
#define HID     64
#define OUT_DIM 10
#define BATCH   4096
#define TSTEPS  128
#define NPAIR   (VOCAB * VOCAB)   // 100 digit pairs
#define PSTRIDE 68                // padded table row stride (dwords) -> bank spread
#define HSTRIDE 72                // padded hbuf row stride (f16): 36 dwords, 16 rows
                                  // map to 16 distinct bank groups (gcd(36,64)=4)
#define WAVES_PER_BLOCK 2
#define ROWS_PER_BLOCK  32        // 16 batch rows per wave

__device__ __forceinline__ float fast_tanh(float x) {
#if __has_builtin(__builtin_amdgcn_tanhf)
    return __builtin_amdgcn_tanhf(x);
#elif __has_builtin(__builtin_amdgcn_tanh_f32)
    return __builtin_amdgcn_tanh_f32(x);
#else
    return tanhf(x);
#endif
}

// ---------------------------------------------------------------------------
// Prep: fold embedding + input projection + bias into ONE 100x64 pair table.
//   P12[d1*10+d2][j] = sum_k E[d1][k]*Wxh[k][j] + E[d2][k]*Wxh[EMB+k][j] + b[j]
// ---------------------------------------------------------------------------
__global__ void rnn_prep(const float* __restrict__ E,
                         const float* __restrict__ Wxh,
                         const float* __restrict__ b,
                         float* __restrict__ P /* [NPAIR*HID] */) {
    int idx = blockIdx.x * blockDim.x + threadIdx.x;
    if (idx >= NPAIR * HID) return;
    int pair = idx / HID, j = idx % HID;
    int d1 = pair / VOCAB, d2 = pair % VOCAB;
    float s = b[j];
    for (int k = 0; k < EMB; ++k) {
        s += E[d1 * EMB + k] * Wxh[k * HID + j];
        s += E[d2 * EMB + k] * Wxh[(EMB + k) * HID + j];
    }
    P[idx] = s;
}

// ---------------------------------------------------------------------------
// Fused recurrent kernel, barrier-free T-loop.
// Each wave owns 16 batch rows end-to-end: per step 8x WMMA recurrence
// (full 16x64 h-update), tanh, wave-private LDS transpose (D-layout ->
// A-fragments), then 2x WMMA logits. A-fragments persist into the next step.
// ---------------------------------------------------------------------------
__global__ __launch_bounds__(32 * WAVES_PER_BLOCK) void rnn_fused(
    const int*   __restrict__ num1,
    const int*   __restrict__ num2,
    const float* __restrict__ Whh,
    const float* __restrict__ Wd,
    const float* __restrict__ bd,
    const float* __restrict__ P,
    float*       __restrict__ out) {

    __shared__ float sP[NPAIR * PSTRIDE];                       // ~27 KB
    __shared__ __align__(16) _Float16 hbuf[WAVES_PER_BLOCK][16 * HSTRIDE];
    __shared__ uint8_t sIdx[ROWS_PER_BLOCK * TSTEPS];           // 4 KB

    const int tid  = threadIdx.x;
    const int lane = tid & 31;
    const int wave = tid >> 5;
    const int nloc = lane & 15;     // M for A-frags / N for B-frags & C tiles
    const int sel  = lane >> 4;     // lane half
    const int b0   = blockIdx.x * ROWS_PER_BLOCK;

    // ---- cooperative staging: pair table (padded) + combined digit indices --
    for (int i = tid; i < NPAIR * HID; i += 32 * WAVES_PER_BLOCK)
        sP[(i >> 6) * PSTRIDE + (i & 63)] = P[i];
    for (int i = tid; i < ROWS_PER_BLOCK * TSTEPS; i += 32 * WAVES_PER_BLOCK)
        sIdx[i] = (uint8_t)(num1[b0 * TSTEPS + i] * VOCAB + num2[b0 * TSTEPS + i]);

    // ---- Whh B-fragments, all 4 N-tiles x 2 K-halves (32x16 f16 B layout) --
    // lane -> N = 16n+nloc; element e -> K = 32s + 16*sel + e
    v16h bhh[4][2];
    for (int n = 0; n < 4; ++n)
        for (int s = 0; s < 2; ++s)
            for (int e = 0; e < 16; ++e) {
                int K = 32 * s + 16 * sel + e;
                bhh[n][s][e] = (_Float16)Whh[K * HID + 16 * n + nloc];
            }

    // ---- Wd fragments (N=10 zero-padded to 16) + bd bias C tile ------------
    v16h bwd[2];
    for (int s = 0; s < 2; ++s)
        for (int e = 0; e < 16; ++e) {
            int K = 32 * s + 16 * sel + e;
            float w = (nloc < OUT_DIM) ? Wd[K * OUT_DIM + nloc] : 0.f;
            bwd[s][e] = (_Float16)w;
        }
    v8f obias;
    {
        float bv = (nloc < OUT_DIM) ? bd[nloc] : 0.f;
        for (int r = 0; r < 8; ++r) obias[r] = bv;
    }

    __syncthreads();   // staging visible; the T-loop itself is barrier-free

    _Float16* myh = hbuf[wave];                 // wave-private transpose scratch
    const int4* hrow = (const int4*)myh + nloc * (HSTRIDE / 8);  // 72 f16 = 9 int4

    // A-fragments carry h_{t-1}; h_0 = 0 lives in registers (no hbuf init).
    union AF { v16h h; int4 q[2]; };
    AF a0, a1;
    {
        int4 z = {0, 0, 0, 0};
        a0.q[0] = z; a0.q[1] = z; a1.q[0] = z; a1.q[1] = z;
    }

    const int idxbase = (wave * 16 + 8 * sel) * TSTEPS;  // my 8 rows in sIdx

    for (int t = 0; t < TSTEPS; ++t) {
        // combined digit-pair table offsets for my 8 rows (2 broadcast addrs/load)
        int off[8];
        for (int r = 0; r < 8; ++r)
            off[r] = (int)sIdx[idxbase + r * TSTEPS + t] * PSTRIDE;

        // h_t = tanh(xp_t + h_{t-1} @ Whh): 4 N-tiles x (2 WMMA, K=64)
        for (int n = 0; n < 4; ++n) {
            const int col = 16 * n + nloc;
            v8f c;
            for (int r = 0; r < 8; ++r) c[r] = sP[off[r] + col];
            c = __builtin_amdgcn_wmma_f32_16x16x32_f16(false, a0.h, false, bhh[n][0],
                                                       (short)0, c, false, false);
            c = __builtin_amdgcn_wmma_f32_16x16x32_f16(false, a1.h, false, bhh[n][1],
                                                       (short)0, c, false, false);
            for (int r = 0; r < 8; ++r)
                myh[(r + 8 * sel) * HSTRIDE + col] = (_Float16)fast_tanh(c[r]);
        }

        // Rebuild A-fragments from h_t (wave-private LDS; s_wait_dscnt only).
        // Contiguous K order (same as B-frags): lane half sel holds K=kb+16*sel..kb+16*sel+15.
        a0.q[0] = hrow[2 * sel];     a0.q[1] = hrow[2 * sel + 1];
        a1.q[0] = hrow[4 + 2 * sel]; a1.q[1] = hrow[5 + 2 * sel];

        // logits_t = h_t @ Wd + bd (reuses the fragments kept for step t+1)
        v8f o = obias;
        o = __builtin_amdgcn_wmma_f32_16x16x32_f16(false, a0.h, false, bwd[0],
                                                   (short)0, o, false, false);
        o = __builtin_amdgcn_wmma_f32_16x16x32_f16(false, a1.h, false, bwd[1],
                                                   (short)0, o, false, false);
        if (nloc < OUT_DIM) {
            for (int r = 0; r < 8; ++r) {
                size_t row = (size_t)(b0 + wave * 16 + r + 8 * sel);
                out[(row * TSTEPS + t) * OUT_DIM + nloc] = o[r];
            }
        }
    }
}

extern "C" void kernel_launch(void* const* d_in, const int* in_sizes, int n_in,
                              void* d_out, int out_size, void* d_ws, size_t ws_size,
                              hipStream_t stream) {
    const int*   num1 = (const int*)  d_in[0];
    const int*   num2 = (const int*)  d_in[1];
    const float* E    = (const float*)d_in[2];
    const float* Wxh  = (const float*)d_in[3];
    const float* Whh  = (const float*)d_in[4];
    const float* b    = (const float*)d_in[5];
    const float* Wd   = (const float*)d_in[6];
    const float* bd   = (const float*)d_in[7];
    float* out = (float*)d_out;
    float* P   = (float*)d_ws;   // NPAIR*HID floats = 25600 B

    rnn_prep<<<(NPAIR * HID + 255) / 256, 256, 0, stream>>>(E, Wxh, b, P);
    rnn_fused<<<BATCH / ROWS_PER_BLOCK, 32 * WAVES_PER_BLOCK, 0, stream>>>(
        num1, num2, Whh, Wd, bd, P, out);
}
